// LearnableUpsampler_75136157876458
// MI455X (gfx1250) — matrix operand order repaired and checked by reading, not compiled, not run on hardware
//
#include <hip/hip_runtime.h>
#include <hip/hip_bf16.h>

// ---------------- problem constants ----------------
#define BB 16
#define DD 256
#define LL 200
#define TT 1000
#define LP 224   // L padded to a multiple of 32 for WMMA K-loops

typedef _Float16 v16h __attribute__((ext_vector_type(16)));
typedef _Float16 v8h  __attribute__((ext_vector_type(8)));
typedef float    v8f  __attribute__((ext_vector_type(8)));

__device__ __forceinline__ float silu_f(float x) { return x / (1.f + __expf(-x)); }

// A-fragment (16x32 f16, M x K): lane holds row M=lane&15; g=lane>>4.
// h=0..7 -> K = k0 + 8g + h ; h=8..15 -> K = k0 + 16 + 8g + (h-8)
__device__ __forceinline__ v16h ld_a16(const _Float16* __restrict__ row, int k0, int g) {
  v8h lo = *(const v8h*)(row + k0 + 8 * g);
  v8h hi = *(const v8h*)(row + k0 + 16 + 8 * g);
  v16h r;
#pragma unroll
  for (int i = 0; i < 8; ++i) { r[i] = lo[i]; r[i + 8] = hi[i]; }
  return r;
}
// B-fragment (32x16 f16, K x N) from Bt stored row-major (N,K):
// lane holds column N=lane&15; h=0..7 -> K = k0 + 8g + h ; h=8..15 -> K = k0 + 16 + 8g + (h-8)
__device__ __forceinline__ v16h ld_b16(const _Float16* __restrict__ row, int k0, int g) {
  v8h lo = *(const v8h*)(row + k0 + 8 * g);
  v8h hi = *(const v8h*)(row + k0 + 16 + 8 * g);
  v16h r;
#pragma unroll
  for (int i = 0; i < 8; ++i) { r[i] = lo[i]; r[i + 8] = hi[i]; }
  return r;
}
#define WMMA_F32_F16(a, b, c) \
  __builtin_amdgcn_wmma_f32_16x16x32_f16(false, (a), false, (b), (short)0, (c), false, false)

// ---------------- workspace layout (bytes) ----------------
static constexpr size_t wal(size_t x) { return (x + 255) & ~(size_t)255; }
constexpr size_t OFF_SK     = 0;                                  // f32 B*L
constexpr size_t OFF_FLEN   = wal(OFF_SK     + (size_t)BB*LL*4);  // i32 B
constexpr size_t OFF_H1W    = wal(OFF_FLEN   + (size_t)BB*4);     // f32 B*D*L
constexpr size_t OFF_H1C    = wal(OFF_H1W    + (size_t)BB*DD*LL*4);
constexpr size_t OFF_HW8    = wal(OFF_H1C    + (size_t)BB*DD*LL*4); // f32 B*L*8
constexpr size_t OFF_HC8    = wal(OFF_HW8    + (size_t)BB*LL*8*4);
constexpr size_t OFF_C      = wal(OFF_HC8    + (size_t)BB*LL*8*4);  // f32 B*T*L*2
constexpr size_t OFF_WHCPRE = wal(OFF_C      + (size_t)BB*TT*LL*2*4); // f32 B*T*256
constexpr size_t OFF_PH16   = wal(OFF_WHCPRE + (size_t)BB*TT*DD*4);   // h B*D*LP
constexpr size_t OFF_PHT16  = wal(OFF_PH16   + (size_t)BB*DD*LP*2);   // h B*LP*D
constexpr size_t OFF_WPW16  = wal(OFF_PHT16  + (size_t)BB*LP*DD*2);   // h 256*256
constexpr size_t OFF_WPC16  = wal(OFF_WPW16  + (size_t)DD*DD*2);
constexpr size_t OFF_LWT16  = wal(OFF_WPC16  + (size_t)DD*DD*2);      // h 256*1024 (Lw^T)
constexpr size_t OFF_WOT16  = wal(OFF_LWT16  + (size_t)DD*1024*2);    // h 512*256  (Wo^T)
constexpr size_t OFF_W16    = wal(OFF_WOT16  + (size_t)512*DD*2);     // h B*4*T*LP
constexpr size_t OFF_WH16   = wal(OFF_W16    + (size_t)BB*4*TT*LP*2); // h B*T*1024
constexpr size_t OFF_WHC16  = wal(OFF_WH16   + (size_t)BB*TT*1024*2); // h B*T*256

// ---------------- output layout (f32 elements) ----------------
constexpr size_t O_MEAN   = 0;                         // B*256*T
constexpr size_t O_LOGSTD = O_MEAN   + (size_t)BB*DD*TT;
constexpr size_t O_FMASK  = O_LOGSTD + (size_t)BB*DD*TT; // B*T
constexpr size_t O_FLEN   = O_FMASK  + (size_t)BB*TT;    // B
constexpr size_t O_W      = O_FLEN   + (size_t)BB;       // B*4*T*L

// ============ K0: cumsum, frame lengths/mask ============
__global__ void k_setup(const float* __restrict__ dur, float* __restrict__ sk,
                        int* __restrict__ flen, float* __restrict__ o_fmask,
                        float* __restrict__ o_flen) {
  const int b = blockIdx.x, tid = threadIdx.x;
  __shared__ float s_sk[LL];
  __shared__ int s_fl;
  if (tid == 0) {
    float run = 0.f;
    for (int l = 0; l < LL; ++l) { s_sk[l] = run; run += dur[b * LL + l]; }
    int fl = (int)rintf(run);
    fl = fl < 0 ? 0 : (fl > TT ? TT : fl);
    s_fl = fl; flen[b] = fl; o_flen[b] = (float)fl;
  }
  __syncthreads();
  for (int l = tid; l < LL; l += blockDim.x) sk[b * LL + l] = s_sk[l];
  for (int t = tid; t < TT; t += blockDim.x)
    o_fmask[b * TT + t] = (t < s_fl) ? 1.f : 0.f;
}

// ============ K1a: f16 pack of phoneme (both layouts, zero-padded) ============
__global__ void k_pack_ph(const float* __restrict__ ph, _Float16* __restrict__ ph16,
                          _Float16* __restrict__ phT16) {
  size_t idx = (size_t)blockIdx.x * 256 + threadIdx.x;
  if (idx >= (size_t)BB * LP * DD) return;
  int d  = (int)(idx % DD);
  int lp = (int)((idx / DD) % LP);
  int b  = (int)(idx / ((size_t)DD * LP));
  float v = (lp < LL) ? ph[((size_t)b * DD + d) * LL + lp] : 0.f;
  phT16[((size_t)b * LP + lp) * DD + d] = (_Float16)v;
  ph16 [((size_t)b * DD + d) * LP + lp] = (_Float16)v;
}

// ============ K1b: f16 pack of weights (Lw^T, Wo^T) ============
__global__ void k_pack_w(const float* __restrict__ Wpw, const float* __restrict__ Wpc,
                         const float* __restrict__ Lw, const float* __restrict__ Wo,
                         _Float16* __restrict__ Wpw16, _Float16* __restrict__ Wpc16,
                         _Float16* __restrict__ LwT16, _Float16* __restrict__ WoT16) {
  int idx = blockIdx.x * 256 + threadIdx.x;
  if (idx < DD * DD) { Wpw16[idx] = (_Float16)Wpw[idx]; Wpc16[idx] = (_Float16)Wpc[idx]; }
  if (idx < DD * 1024) { int n = idx / 1024, j = idx % 1024; LwT16[idx] = (_Float16)Lw[(size_t)j * DD + n]; }
  if (idx < 512 * DD) { int n = idx / DD, k = idx % DD; WoT16[idx] = (_Float16)Wo[(size_t)k * 512 + n]; }
}

// ============ K2: projection GEMM (both branches), 2x2 register-blocked ============
__global__ void k_gemm_proj(const _Float16* __restrict__ Aw, const _Float16* __restrict__ Ac,
                            const _Float16* __restrict__ phT,
                            const float* __restrict__ bpw, const float* __restrict__ bpc,
                            float* __restrict__ h1w, float* __restrict__ h1c) {
  const int lane = threadIdx.x & 31;
  const int ntp = blockIdx.x;           // 7 pairs -> n tiles 2*ntp, 2*ntp+1 (<= 13)
  const int mtp = blockIdx.y;           // 8 pairs over D
  const int bz = blockIdx.z;            // B*2
  const int b = bz >> 1, br = bz & 1;
  const _Float16* A = br ? Ac : Aw;
  const float* bias = br ? bpc : bpw;
  float* out = (br ? h1c : h1w) + (size_t)b * DD * LL;
  const int g = lane >> 4, r = lane & 15;
  const _Float16* a0row = A + (size_t)(mtp * 32 + r) * DD;
  const _Float16* a1row = a0row + (size_t)16 * DD;
  const _Float16* b0row = phT + ((size_t)b * LP + ntp * 32 + r) * DD;
  const _Float16* b1row = b0row + (size_t)16 * DD;
  v8f acc00 = {}, acc01 = {}, acc10 = {}, acc11 = {};
  for (int k0 = 0; k0 < DD; k0 += 32) {
    v16h a0 = ld_a16(a0row, k0, g);
    v16h a1 = ld_a16(a1row, k0, g);
    v16h b0 = ld_b16(b0row, k0, g);
    v16h b1 = ld_b16(b1row, k0, g);
    acc00 = WMMA_F32_F16(a0, b0, acc00);
    acc01 = WMMA_F32_F16(a0, b1, acc01);
    acc10 = WMMA_F32_F16(a1, b0, acc10);
    acc11 = WMMA_F32_F16(a1, b1, acc11);
  }
#pragma unroll
  for (int i = 0; i < 4; ++i) {
    const int mt = mtp * 2 + (i >> 1);
    const int nt = ntp * 2 + (i & 1);
    const v8f acc = (i == 0) ? acc00 : (i == 1) ? acc01 : (i == 2) ? acc10 : acc11;
    const int col = nt * 16 + r;
    if (col < LL) {
#pragma unroll
      for (int v = 0; v < 8; ++v) {
        const int rowm = mt * 16 + v + 8 * g;
        out[(size_t)rowm * LL + col] = acc[v] + bias[rowm];
      }
    }
  }
}

// ============ K3: depth-256 -> 8ch conv(K=3) + silu + pm mask ============
__global__ void k_conv(const float* __restrict__ h1w, const float* __restrict__ h1c,
                       const float* __restrict__ dur,
                       const float* __restrict__ Cw, const float* __restrict__ cbw,
                       const float* __restrict__ Cc, const float* __restrict__ cbc,
                       float* __restrict__ hw8, float* __restrict__ hc8) {
  const int bl = blockIdx.x;
  const int b = bl / LL, l = bl % LL;
  const int tid = threadIdx.x;  // 0..255 == input channel d
  const bool pmv = dur[b * LL + l] > 0.f;
  __shared__ float red[256 * 8];
  for (int br = 0; br < 2; ++br) {
    const float* h1 = (br ? h1c : h1w) + ((size_t)b * DD + tid) * LL;
    const float* C  = br ? Cc : Cw;
    const float* cb = br ? cbc : cbw;
    const float xm = (l > 0)      ? h1[l - 1] : 0.f;
    const float x0 = h1[l];
    const float xp = (l < LL - 1) ? h1[l + 1] : 0.f;
#pragma unroll
    for (int o = 0; o < 8; ++o) {
      const float* c3 = C + ((size_t)(o * DD + tid)) * 3;
      red[tid * 8 + o] = xm * c3[0] + x0 * c3[1] + xp * c3[2];
    }
    __syncthreads();
    for (int s = 128; s > 0; s >>= 1) {
      if (tid < s) {
#pragma unroll
        for (int o = 0; o < 8; ++o) red[tid * 8 + o] += red[(tid + s) * 8 + o];
      }
      __syncthreads();
    }
    if (tid < 8) {
      float z = red[tid] + cb[tid];
      (br ? hc8 : hw8)[((size_t)(b * LL + l)) * 8 + tid] = pmv ? silu_f(z) : 0.f;
    }
    __syncthreads();
  }
}

// ============ K4: MLPs + masked softmax over L -> w (f32 out + f16 packed), c ====
__global__ void k_wmlp(const float* __restrict__ dur, const float* __restrict__ sk,
                       const int* __restrict__ flen,
                       const float* __restrict__ hw8, const float* __restrict__ hc8,
                       const float* __restrict__ Mw, const float* __restrict__ mbw,
                       const float* __restrict__ Mc, const float* __restrict__ mbc,
                       float* __restrict__ w_out, _Float16* __restrict__ w16,
                       float* __restrict__ c_out) {
  const int bt = blockIdx.x;
  const int b = bt / TT, t = bt % TT;
  const int lane = threadIdx.x;
  float mw[40], mb4[4], mc[20], mb2[2];
#pragma unroll
  for (int i = 0; i < 40; ++i) mw[i] = Mw[i];
#pragma unroll
  for (int i = 0; i < 4; ++i) mb4[i] = mbw[i];
#pragma unroll
  for (int i = 0; i < 20; ++i) mc[i] = Mc[i];
#pragma unroll
  for (int i = 0; i < 2; ++i) mb2[i] = mbc[i];
  const float tv = (float)(t + 1);
  const bool on = t < flen[b];
  float logit[7][4];
  float lmax[4] = {-1e30f, -1e30f, -1e30f, -1e30f};
#pragma unroll
  for (int ci = 0; ci < 7; ++ci) {
    const int l = ci * 32 + lane;
#pragma unroll
    for (int q = 0; q < 4; ++q) logit[ci][q] = -1e30f;
    if (l < LL) {
      const float dl = dur[b * LL + l];
      const float S = tv - sk[b * LL + l];
      const float E = dl;
      const float* hp = hw8 + ((size_t)(b * LL + l)) * 8;
      float hv[8];
#pragma unroll
      for (int j = 0; j < 8; ++j) hv[j] = hp[j];
      if (dl > 0.f) {
#pragma unroll
        for (int q = 0; q < 4; ++q) {
          float z = S * mw[q] + E * mw[4 + q] + mb4[q];
#pragma unroll
          for (int j = 0; j < 8; ++j) z += hv[j] * mw[(2 + j) * 4 + q];
          const float s = silu_f(z);
          logit[ci][q] = s;
          lmax[q] = fmaxf(lmax[q], s);
        }
      }
      const float* hcp = hc8 + ((size_t)(b * LL + l)) * 8;
      float hcv[8];
#pragma unroll
      for (int j = 0; j < 8; ++j) hcv[j] = hcp[j];
#pragma unroll
      for (int p = 0; p < 2; ++p) {
        float z = S * mc[p] + E * mc[2 + p] + mb2[p];
#pragma unroll
        for (int j = 0; j < 8; ++j) z += hcv[j] * mc[(2 + j) * 2 + p];
        c_out[(((size_t)b * TT + t) * LL + l) * 2 + p] = silu_f(z);
      }
    }
  }
#pragma unroll
  for (int q = 0; q < 4; ++q)
    for (int off = 16; off > 0; off >>= 1)
      lmax[q] = fmaxf(lmax[q], __shfl_xor(lmax[q], off, 32));
  float esum[4] = {0.f, 0.f, 0.f, 0.f};
#pragma unroll
  for (int ci = 0; ci < 7; ++ci)
#pragma unroll
    for (int q = 0; q < 4; ++q) {
      const float e = __expf(logit[ci][q] - lmax[q]);  // 0 for masked / padded lanes
      logit[ci][q] = e;
      esum[q] += e;
    }
#pragma unroll
  for (int q = 0; q < 4; ++q)
    for (int off = 16; off > 0; off >>= 1) esum[q] += __shfl_xor(esum[q], off, 32);
  float inv[4];
#pragma unroll
  for (int q = 0; q < 4; ++q) inv[q] = 1.f / esum[q];
#pragma unroll
  for (int ci = 0; ci < 7; ++ci) {
    const int l = ci * 32 + lane;
    if (l < LL) {
#pragma unroll
      for (int q = 0; q < 4; ++q) {
        const float wv = on ? logit[ci][q] * inv[q] : 0.f;
        const size_t base = (size_t)(b * 4 + q) * TT + t;
        w_out[base * LL + l] = wv;
        w16[base * LP + l] = (_Float16)wv;
      }
    }
  }
  if (lane < LP - LL) {
#pragma unroll
    for (int q = 0; q < 4; ++q)
      w16[((size_t)(b * 4 + q) * TT + t) * LP + LL + lane] = (_Float16)0.f;
  }
}

// ============ K5: wh GEMM (T*4 x L x D) per batch, 2x2 blocked, f16 out =========
__global__ void k_gemm_wh(const _Float16* __restrict__ w16, const _Float16* __restrict__ ph16,
                          _Float16* __restrict__ wh16) {
  const int lane = threadIdx.x & 31;
  const int ntp = blockIdx.x;   // 8 pairs over D (16 tiles)
  const int mtp = blockIdx.y;   // 125 pairs over T*4 (250 tiles)
  const int b  = blockIdx.z;
  const int g = lane >> 4, r = lane & 15;
  const int m0 = mtp * 32 + r;            // row of first m-tile
  const int m1 = m0 + 16;
  const _Float16* a0row = w16 + ((size_t)(b * 4 + (m0 & 3)) * TT + (m0 >> 2)) * LP;
  const _Float16* a1row = w16 + ((size_t)(b * 4 + (m1 & 3)) * TT + (m1 >> 2)) * LP;
  const _Float16* b0row = ph16 + ((size_t)b * DD + ntp * 32 + r) * LP;
  const _Float16* b1row = b0row + (size_t)16 * LP;
  v8f acc00 = {}, acc01 = {}, acc10 = {}, acc11 = {};
  for (int k0 = 0; k0 < LP; k0 += 32) {
    v16h a0 = ld_a16(a0row, k0, g);
    v16h a1 = ld_a16(a1row, k0, g);
    v16h b0 = ld_b16(b0row, k0, g);
    v16h b1 = ld_b16(b1row, k0, g);
    acc00 = WMMA_F32_F16(a0, b0, acc00);
    acc01 = WMMA_F32_F16(a0, b1, acc01);
    acc10 = WMMA_F32_F16(a1, b0, acc10);
    acc11 = WMMA_F32_F16(a1, b1, acc11);
  }
#pragma unroll
  for (int i = 0; i < 4; ++i) {
    const int mt = mtp * 2 + (i >> 1);
    const int nt = ntp * 2 + (i & 1);
    const v8f acc = (i == 0) ? acc00 : (i == 1) ? acc01 : (i == 2) ? acc10 : acc11;
#pragma unroll
    for (int v = 0; v < 8; ++v) {
      const int mm = mt * 16 + v + 8 * g;
      const int tt = mm >> 2, qq = mm & 3;
      wh16[((size_t)b * TT + tt) * 1024 + qq * DD + nt * 16 + r] = (_Float16)acc[v];
    }
  }
}

// ============ K6: whc_pre = wh @ Lw (T x 1024 x 256), 2x2 blocked ============
__global__ void k_gemm_whcpre(const _Float16* __restrict__ wh16, const _Float16* __restrict__ LwT,
                              float* __restrict__ whcpre) {
  const int lane = threadIdx.x & 31;
  const int ntp = blockIdx.x;   // 8 pairs over 256
  const int mtp = blockIdx.y;   // 32 pairs over T (64 tiles, clamped)
  const int b  = blockIdx.z;
  const int g = lane >> 4, r = lane & 15;
  const int t0 = mtp * 32 + r;
  const int t1 = t0 + 16;
  const int t0c = t0 < TT ? t0 : TT - 1;
  const int t1c = t1 < TT ? t1 : TT - 1;
  const _Float16* a0row = wh16 + ((size_t)b * TT + t0c) * 1024;
  const _Float16* a1row = wh16 + ((size_t)b * TT + t1c) * 1024;
  const _Float16* b0row = LwT + (size_t)(ntp * 32 + r) * 1024;
  const _Float16* b1row = b0row + (size_t)16 * 1024;
  v8f acc00 = {}, acc01 = {}, acc10 = {}, acc11 = {};
  for (int k0 = 0; k0 < 1024; k0 += 32) {
    v16h a0 = ld_a16(a0row, k0, g);
    v16h a1 = ld_a16(a1row, k0, g);
    v16h b0 = ld_b16(b0row, k0, g);
    v16h b1 = ld_b16(b1row, k0, g);
    acc00 = WMMA_F32_F16(a0, b0, acc00);
    acc01 = WMMA_F32_F16(a0, b1, acc01);
    acc10 = WMMA_F32_F16(a1, b0, acc10);
    acc11 = WMMA_F32_F16(a1, b1, acc11);
  }
#pragma unroll
  for (int i = 0; i < 4; ++i) {
    const int mt = mtp * 2 + (i >> 1);
    const int nt = ntp * 2 + (i & 1);
    const v8f acc = (i == 0) ? acc00 : (i == 1) ? acc01 : (i == 2) ? acc10 : acc11;
    const int col = nt * 16 + r;
#pragma unroll
    for (int v = 0; v < 8; ++v) {
      const int tr = mt * 16 + v + 8 * g;
      if (tr < TT) whcpre[((size_t)b * TT + tr) * DD + col] = acc[v];
    }
  }
}

// ============ K7: wc contraction + Lc + biases + frame mask -> whc f16 ============
__global__ void k_wc(const float* __restrict__ w_out, const float* __restrict__ c_out,
                     const float* __restrict__ whcpre, const float* __restrict__ Lc,
                     const float* __restrict__ lbw, const float* __restrict__ lbc,
                     const int* __restrict__ flen, _Float16* __restrict__ whc16) {
  const int bt = blockIdx.x;
  const int b = bt / TT, t = bt % TT;
  const int lane = threadIdx.x;
  float acc8[8] = {0.f, 0.f, 0.f, 0.f, 0.f, 0.f, 0.f, 0.f};
#pragma unroll
  for (int ci = 0; ci < 7; ++ci) {
    const int l = ci * 32 + lane;
    if (l < LL) {
      float wv[4];
#pragma unroll
      for (int q = 0; q < 4; ++q)
        wv[q] = w_out[((size_t)(b * 4 + q) * TT + t) * LL + l];
      const float* cp = c_out + (((size_t)b * TT + t) * LL + l) * 2;
      const float c0 = cp[0], c1 = cp[1];
#pragma unroll
      for (int q = 0; q < 4; ++q) { acc8[q * 2] += wv[q] * c0; acc8[q * 2 + 1] += wv[q] * c1; }
    }
  }
#pragma unroll
  for (int j = 0; j < 8; ++j)
    for (int off = 16; off > 0; off >>= 1) acc8[j] += __shfl_xor(acc8[j], off, 32);
  const bool on = t < flen[b];
#pragma unroll
  for (int i = 0; i < 8; ++i) {
    const int dp = lane + 32 * i;
    float v = 0.f;
    if (on) {
      v = whcpre[((size_t)b * TT + t) * DD + dp] + lbw[dp] + lbc[dp];
#pragma unroll
      for (int j = 0; j < 8; ++j) v += acc8[j] * Lc[j * DD + dp];
    }
    whc16[((size_t)b * TT + t) * DD + dp] = (_Float16)v;
  }
}

// ============ K8: out = whc @ Wo + bo, 2x2 blocked, transposed split store =======
__global__ void k_gemm_out(const _Float16* __restrict__ whc16, const _Float16* __restrict__ WoT,
                           const float* __restrict__ bo,
                           float* __restrict__ o_mean, float* __restrict__ o_logstd) {
  const int lane = threadIdx.x & 31;
  const int ntp = blockIdx.x;   // 16 pairs over 512
  const int mtp = blockIdx.y;   // 32 pairs over T (64 tiles, clamped)
  const int b  = blockIdx.z;
  const int g = lane >> 4, r = lane & 15;
  const int t0 = mtp * 32 + r;
  const int t1 = t0 + 16;
  const int t0c = t0 < TT ? t0 : TT - 1;
  const int t1c = t1 < TT ? t1 : TT - 1;
  const _Float16* a0row = whc16 + ((size_t)b * TT + t0c) * DD;
  const _Float16* a1row = whc16 + ((size_t)b * TT + t1c) * DD;
  const _Float16* b0row = WoT + (size_t)(ntp * 32 + r) * DD;
  const _Float16* b1row = b0row + (size_t)16 * DD;
  v8f acc00 = {}, acc01 = {}, acc10 = {}, acc11 = {};
  for (int k0 = 0; k0 < DD; k0 += 32) {
    v16h a0 = ld_a16(a0row, k0, g);
    v16h a1 = ld_a16(a1row, k0, g);
    v16h b0 = ld_b16(b0row, k0, g);
    v16h b1 = ld_b16(b1row, k0, g);
    acc00 = WMMA_F32_F16(a0, b0, acc00);
    acc01 = WMMA_F32_F16(a0, b1, acc01);
    acc10 = WMMA_F32_F16(a1, b0, acc10);
    acc11 = WMMA_F32_F16(a1, b1, acc11);
  }
#pragma unroll
  for (int i = 0; i < 4; ++i) {
    const int mt = mtp * 2 + (i >> 1);
    const int nt = ntp * 2 + (i & 1);
    const v8f acc = (i == 0) ? acc00 : (i == 1) ? acc01 : (i == 2) ? acc10 : acc11;
    const int n = nt * 16 + r;
#pragma unroll
    for (int v = 0; v < 8; ++v) {
      const int tr = mt * 16 + v + 8 * g;
      if (tr < TT) {
        const float val = acc[v] + bo[n];
        if (n < DD) o_mean[((size_t)b * DD + n) * TT + tr] = val;
        else        o_logstd[((size_t)b * DD + (n - DD)) * TT + tr] = val;
      }
    }
  }
}

// ================================================================
extern "C" void kernel_launch(void* const* d_in, const int* in_sizes, int n_in,
                              void* d_out, int out_size, void* d_ws, size_t ws_size,
                              hipStream_t stream) {
  (void)in_sizes; (void)n_in; (void)out_size; (void)ws_size;
  const float* dur = (const float*)d_in[0];
  const float* ph  = (const float*)d_in[1];
  // d_in[2] = phoneme_mask (bool) — recovered as dur > 0 instead
  const float* Wpw = (const float*)d_in[3];
  const float* bpw = (const float*)d_in[4];
  const float* Cw  = (const float*)d_in[5];
  const float* cbw = (const float*)d_in[6];
  const float* Mw  = (const float*)d_in[7];
  const float* mbw = (const float*)d_in[8];
  const float* Lw  = (const float*)d_in[9];
  const float* lbw = (const float*)d_in[10];
  const float* Wpc = (const float*)d_in[11];
  const float* bpc = (const float*)d_in[12];
  const float* Cc  = (const float*)d_in[13];
  const float* cbc = (const float*)d_in[14];
  const float* Mc  = (const float*)d_in[15];
  const float* mbc = (const float*)d_in[16];
  const float* Lc  = (const float*)d_in[17];
  const float* lbc = (const float*)d_in[18];
  const float* Wo  = (const float*)d_in[19];
  const float* bo  = (const float*)d_in[20];

  char* ws = (char*)d_ws;
  float*     sk     = (float*)(ws + OFF_SK);
  int*       flen   = (int*)(ws + OFF_FLEN);
  float*     h1w    = (float*)(ws + OFF_H1W);
  float*     h1c    = (float*)(ws + OFF_H1C);
  float*     hw8    = (float*)(ws + OFF_HW8);
  float*     hc8    = (float*)(ws + OFF_HC8);
  float*     c_buf  = (float*)(ws + OFF_C);
  float*     whcpre = (float*)(ws + OFF_WHCPRE);
  _Float16*  ph16   = (_Float16*)(ws + OFF_PH16);
  _Float16*  phT16  = (_Float16*)(ws + OFF_PHT16);
  _Float16*  Wpw16  = (_Float16*)(ws + OFF_WPW16);
  _Float16*  Wpc16  = (_Float16*)(ws + OFF_WPC16);
  _Float16*  LwT16  = (_Float16*)(ws + OFF_LWT16);
  _Float16*  WoT16  = (_Float16*)(ws + OFF_WOT16);
  _Float16*  w16    = (_Float16*)(ws + OFF_W16);
  _Float16*  wh16   = (_Float16*)(ws + OFF_WH16);
  _Float16*  whc16  = (_Float16*)(ws + OFF_WHC16);

  float* out      = (float*)d_out;
  float* o_mean   = out + O_MEAN;
  float* o_logstd = out + O_LOGSTD;
  float* o_fmask  = out + O_FMASK;
  float* o_flen   = out + O_FLEN;
  float* o_w      = out + O_W;

  k_setup<<<BB, 256, 0, stream>>>(dur, sk, flen, o_fmask, o_flen);

  {
    const size_t n = (size_t)BB * LP * DD;
    k_pack_ph<<<(unsigned)((n + 255) / 256), 256, 0, stream>>>(ph, ph16, phT16);
  }
  k_pack_w<<<(DD * 1024 + 255) / 256, 256, 0, stream>>>(Wpw, Wpc, Lw, Wo,
                                                        Wpw16, Wpc16, LwT16, WoT16);

  k_gemm_proj<<<dim3(7, 8, BB * 2), 32, 0, stream>>>(Wpw16, Wpc16, phT16, bpw, bpc, h1w, h1c);

  k_conv<<<BB * LL, 256, 0, stream>>>(h1w, h1c, dur, Cw, cbw, Cc, cbc, hw8, hc8);

  k_wmlp<<<BB * TT, 32, 0, stream>>>(dur, sk, flen, hw8, hc8, Mw, mbw, Mc, mbc,
                                     o_w, w16, c_buf);

  k_gemm_wh<<<dim3(8, 125, BB), 32, 0, stream>>>(w16, ph16, wh16);

  k_gemm_whcpre<<<dim3(8, 32, BB), 32, 0, stream>>>(wh16, LwT16, whcpre);

  k_wc<<<BB * TT, 32, 0, stream>>>(o_w, c_buf, whcpre, Lc, lbw, lbc, flen, whc16);

  k_gemm_out<<<dim3(16, 32, BB), 32, 0, stream>>>(whc16, WoT16, bo, o_mean, o_logstd);
}
